// LogSpaceTripleRCell_16801912062686
// MI455X (gfx1250) — matrix operand order repaired and checked by reading, not compiled, not run on hardware
//
#include <hip/hip_runtime.h>
#include <hip/hip_bf16.h>
#include <stdint.h>

// ---------------- problem constants (reference: T=512, B=32, D=2048) -------
#define TT   512
#define BB   32
#define DD   2048
#define BD   (BB*DD)                 // 65536
#define TBD  ((size_t)TT*BD)         // 33554432
#define TP1BD ((size_t)(TT+1)*BD)    // 33619968

typedef __attribute__((ext_vector_type(8)))  __bf16 v8bf;
typedef __attribute__((ext_vector_type(16))) __bf16 v16bf;
typedef __attribute__((ext_vector_type(8)))  float  v8f;

// ---------------- WMMA helpers ---------------------------------------------
__device__ __forceinline__ v16bf load_a16(const __bf16* row, int k0, int half) {
    // contiguous 16-bit A layout (matches B): lanes0-15 K=[0..15], lanes16-31 K=[16..31]
    v8bf lo = *(const v8bf*)(row + k0 + 16*half);
    v8bf hi = *(const v8bf*)(row + k0 + 16*half + 8);
    return __builtin_shufflevector(lo, hi, 0,1,2,3,4,5,6,7,8,9,10,11,12,13,14,15);
}

__device__ __forceinline__ v8f wmma_bf16(v16bf a, v16bf b, v8f c) {
    return __builtin_amdgcn_wmma_f32_16x16x32_bf16(false, a, false, b, (short)0, c,
                                                   false, false);
}

// ---------------- small utility kernels ------------------------------------
__global__ void cvt_bf16_kernel(const float* __restrict__ src, __bf16* __restrict__ dst, int n) {
    int i = blockIdx.x*256 + threadIdx.x;
    if (i < n) dst[i] = (__bf16)src[i];
}

__global__ void u_init_kernel(float* __restrict__ u) {
    int i = blockIdx.x*256 + threadIdx.x;
    u[i] = cosf((float)i * 0.7f + 0.3f);
}

// y = W^T u   (thread per column, coalesced row-major reads)
__global__ void matvecT_kernel(const float* __restrict__ W, const float* __restrict__ u,
                               float* __restrict__ y) {
    int col = blockIdx.x*256 + threadIdx.x;
    float s = 0.f;
    for (int j = 0; j < DD; ++j) s = fmaf(W[(size_t)j*DD + col], u[j], s);
    y[col] = s;
}

// y = W v   (block per row, LDS reduction)
__global__ void matvec_kernel(const float* __restrict__ W, const float* __restrict__ v,
                              float* __restrict__ y) {
    __shared__ float red[256];
    int row = blockIdx.x;
    float s = 0.f;
    for (int j = threadIdx.x; j < DD; j += 256) s = fmaf(W[(size_t)row*DD + j], v[j], s);
    red[threadIdx.x] = s; __syncthreads();
    for (int st = 128; st > 0; st >>= 1) {
        if (threadIdx.x < st) red[threadIdx.x] += red[threadIdx.x+st];
        __syncthreads();
    }
    if (threadIdx.x == 0) y[row] = red[0];
}

// x /= (||x|| + eps), single block
__global__ void normalize_kernel(float* __restrict__ x, float eps) {
    __shared__ float red[256];
    __shared__ float inv;
    float s = 0.f;
    for (int i = threadIdx.x; i < DD; i += 256) { float t = x[i]; s += t*t; }
    red[threadIdx.x] = s; __syncthreads();
    for (int st = 128; st > 0; st >>= 1) {
        if (threadIdx.x < st) red[threadIdx.x] += red[threadIdx.x+st];
        __syncthreads();
    }
    if (threadIdx.x == 0) inv = 1.0f / (sqrtf(red[0]) + eps);
    __syncthreads();
    for (int i = threadIdx.x; i < DD; i += 256) x[i] *= inv;
}

// sigma = |a . b|, single block
__global__ void dot_abs_kernel(const float* __restrict__ a, const float* __restrict__ b,
                               float* __restrict__ out) {
    __shared__ float red[256];
    float s = 0.f;
    for (int i = threadIdx.x; i < DD; i += 256) s = fmaf(a[i], b[i], s);
    red[threadIdx.x] = s; __syncthreads();
    for (int st = 128; st > 0; st >>= 1) {
        if (threadIdx.x < st) red[threadIdx.x] += red[threadIdx.x+st];
        __syncthreads();
    }
    if (threadIdx.x == 0) out[0] = fabsf(red[0]);
}

// Rh_bf16 = bf16( R_h * 0.9/(sigma+1e-8) )
__global__ void scale_cvt_rh_kernel(const float* __restrict__ Rh, const float* __restrict__ sigma,
                                    __bf16* __restrict__ dst) {
    size_t i = (size_t)blockIdx.x*256 + threadIdx.x;
    float s = 0.9f / (sigma[0] + 1e-8f);
    dst[i] = (__bf16)(Rh[i] * s);
}

// h0 = sign_h0*exp(log_h0); copy t=0 slices of log/sign outputs; reset barrier
__global__ void init_state_kernel(const float* __restrict__ log_h0, const float* __restrict__ sign_h0,
                                  float* __restrict__ hF0, __bf16* __restrict__ hA0,
                                  float* __restrict__ log_out0, float* __restrict__ sign_out0,
                                  unsigned* __restrict__ sync) {
    int i = blockIdx.x*256 + threadIdx.x;
    if (i < BD) {
        float h = sign_h0[i] * expf(log_h0[i]);   // exp(-1e30) -> 0
        hF0[i] = h;
        hA0[i] = (__bf16)h;
        log_out0[i]  = log_h0[i];
        sign_out0[i] = sign_h0[i];
    }
    if (blockIdx.x == 0 && threadIdx.x == 0) { sync[0] = 0u; sync[1] = 0u; }
}

// ---------------- hoisted big GEMM:  Out[M,N] = A[M,K] * W[N,K]^T + bias ----
// M=16384, N=K=2048, bf16 inputs, f32 out. Each wave: 1 Mtile x 4 Ntiles.
__global__ __launch_bounds__(256) void gemm_pre_kernel(const __bf16* __restrict__ Abf,
                                                       const __bf16* __restrict__ Wbf,
                                                       const float* __restrict__ bias,
                                                       float* __restrict__ Out) {
    int wg   = (blockIdx.x*256 + threadIdx.x) >> 5;   // global wave id
    int lane = threadIdx.x & 31;
    int l16  = lane & 15;
    int half = lane >> 4;
    int mtile = wg >> 5;          // 1024 M tiles
    int ngrp  = wg & 31;          // 32 groups of 4 N tiles

    const __bf16* Arow = Abf + (size_t)(mtile*16 + l16)*DD;
    v8f c[4] = {{}, {}, {}, {}};

    for (int k0 = 0; k0 < DD; k0 += 32) {
        v16bf a = load_a16(Arow, k0, half);
        #pragma unroll
        for (int j = 0; j < 4; ++j) {
            const __bf16* Brow = Wbf + (size_t)((ngrp*4 + j)*16 + l16)*DD;
            v16bf b = *(const v16bf*)(Brow + k0 + 16*half);
            c[j] = wmma_bf16(a, b, c[j]);
        }
    }
    #pragma unroll
    for (int j = 0; j < 4; ++j) {
        int nj = (ngrp*4 + j)*16 + l16;
        float bb = bias[nj];
        #pragma unroll
        for (int e = 0; e < 8; ++e) {
            int mm = mtile*16 + e + 8*half;
            Out[(size_t)mm*DD + nj] = c[j][e] + bb;
        }
    }
}

// ---------------- persistent recurrent scan --------------------------------
__device__ __forceinline__ void grid_sync(unsigned* cnt, volatile unsigned* sense, unsigned nb) {
    __threadfence();
    __syncthreads();
    if (threadIdx.x == 0) {
        unsigned s = *sense;
        if (atomicAdd(cnt, 1u) == nb - 1u) {
            *cnt = 0u;
            __threadfence();
            atomicAdd((unsigned*)sense, 1u);
        } else {
            while (*sense == s) { __builtin_amdgcn_s_sleep(1); }
        }
        __threadfence();
    }
    __syncthreads();
}

// 32 blocks x 256 threads (8 waves). Wave w: mtile = w&1, ntile = blk*4 + (w>>1).
// Per-block LDS: 4 Ntiles of Rh_eff (64 rows x 2048 bf16 = 256 KB) cached ONCE
// for all 512 steps (WGP has 320 KB LDS). Filled with CDNA5 async copies
// (GLOBAL_LOAD_ASYNC_TO_LDS_B128, tracked by ASYNCcnt) -- no VGPR round trip.
__global__ __launch_bounds__(256) void scan_kernel(
        __bf16* __restrict__ hA0, __bf16* __restrict__ hA1,
        float*  __restrict__ hF0, float*  __restrict__ hF1,
        const __bf16* __restrict__ RhBf,
        float* OUT, float* HLIN,                 // staged P / Q, overwritten in place
        float* __restrict__ LOG1, float* __restrict__ SGN1,   // log_h[1:], sign_h[1:]
        const float* __restrict__ x, const float* __restrict__ b_gate,
        unsigned* __restrict__ sync)
{
    extern __shared__ char smem_raw[];
    __bf16* ldsB = (__bf16*)smem_raw;            // [64][2048]

    {   // async fill of the weight cache: 16384 x 16B chunks, direct to LDS
        const char* srcB = (const char*)(RhBf + (size_t)blockIdx.x * 64 * DD);
        unsigned ldsbase = (unsigned)(size_t)(void*)ldsB;   // low 32 bits = LDS offset
        for (int i = threadIdx.x; i < 64*DD*2/16; i += 256) {
            unsigned           ldst = ldsbase + (unsigned)i*16u;
            unsigned long long gsrc = (unsigned long long)(size_t)(srcB + (size_t)i*16);
            asm volatile("global_load_async_to_lds_b128 %0, %1, off"
                         :: "v"(ldst), "v"(gsrc) : "memory");
        }
        asm volatile("s_wait_asynccnt 0x0" ::: "memory");
    }
    __syncthreads();

    const int lane  = threadIdx.x & 31;
    const int w     = threadIdx.x >> 5;
    const int l16   = lane & 15;
    const int half  = lane >> 4;
    const int pair  = w >> 1;                    // 0..3 -> which cached ntile
    const int mtile = w & 1;                     // 0..1
    const int ntile = blockIdx.x*4 + pair;
    const int nn    = ntile*16 + l16;
    const __bf16* Blane = ldsB + (size_t)(pair*16 + l16)*DD;
    const float bg = b_gate[nn];

    for (int t = 0; t < TT; ++t) {
        const __bf16* hA  = (t & 1) ? hA1 : hA0;
        const float*  hF  = (t & 1) ? hF1 : hF0;
        __bf16*       hAn = (t & 1) ? hA0 : hA1;
        float*        hFn = (t & 1) ? hF0 : hF1;

        const __bf16* Arow = hA + (size_t)(mtile*16 + l16)*DD;

        // software-pipelined K loop: prefetch the next 64-K chunk (regs) while
        // the current two WMMAs execute; prefetch index clamped on last iter.
        v8f c0 = {}, c1 = {};
        v16bf a0 = load_a16(Arow, 0, half);
        v16bf b0 = *(const v16bf*)(Blane + 16*half);
        v16bf a1 = load_a16(Arow, 32, half);
        v16bf b1 = *(const v16bf*)(Blane + 32 + 16*half);
        for (int k0 = 0; k0 < DD; k0 += 64) {
            int kn = (k0 + 64 < DD) ? (k0 + 64) : 0;   // safe dummy on last iter
            v16bf a2 = load_a16(Arow, kn, half);
            v16bf b2 = *(const v16bf*)(Blane + kn + 16*half);
            v16bf a3 = load_a16(Arow, kn + 32, half);
            v16bf b3 = *(const v16bf*)(Blane + kn + 32 + 16*half);
            c0 = wmma_bf16(a0, b0, c0);
            c1 = wmma_bf16(a1, b1, c1);
            a0 = a2; b0 = b2; a1 = a3; b1 = b3;
        }
        v8f c = c0 + c1;

        const size_t base = (size_t)t * BD;
        #pragma unroll
        for (int e = 0; e < 8; ++e) {
            int bm  = mtile*16 + e + 8*half;     // batch index 0..31
            int idx = bm*DD + nn;
            float hp   = hF[idx];
            float v    = c[e] + OUT[base + idx];           // P already has +b
            float cand = tanhf(v);
            float q    = HLIN[base + idx];                 // Q already has +b_delta
            float dlt  = 1.0f / (1.0f + expf(-q));
            float hn   = (1.0f - dlt)*hp + dlt*cand;
            float z    = hn + x[base + idx] + bg;
            float gate = z / (1.0f + expf(-z));            // silu
            OUT [base + idx] = hn * gate;
            HLIN[base + idx] = hn;
            LOG1[base + idx] = logf(fmaxf(fabsf(hn), 1e-30f));
            SGN1[base + idx] = (hn >= 0.0f) ? 1.0f : -1.0f;
            hFn[idx] = hn;
            hAn[idx] = (__bf16)hn;
        }
        grid_sync(sync, sync + 1, gridDim.x);
    }
}

// ---------------- workspace layout (bytes) ----------------------------------
#define WS_XBF    ((size_t)0)                    // 67108864  x as bf16
#define WS_RXBF   ((size_t)67108864)             // 8388608   R_x bf16
#define WS_WDBF   ((size_t)75497472)             // 8388608   W_delta bf16
#define WS_RHBF   ((size_t)83886080)             // 8388608   R_h_eff bf16
#define WS_HF     ((size_t)92274688)             // 2 * 262144 f32 state
#define WS_HBF    ((size_t)92798976)             // 2 * 131072 bf16 state
#define WS_U      ((size_t)93061120)
#define WS_V      ((size_t)93069312)
#define WS_W      ((size_t)93077504)
#define WS_SIGMA  ((size_t)93085696)
#define WS_SYNC   ((size_t)93085712)

extern "C" void kernel_launch(void* const* d_in, const int* in_sizes, int n_in,
                              void* d_out, int out_size, void* d_ws, size_t ws_size,
                              hipStream_t stream) {
    const float* x       = (const float*)d_in[0];
    const float* log_h0  = (const float*)d_in[1];
    const float* sign_h0 = (const float*)d_in[2];
    const float* R_h     = (const float*)d_in[3];
    const float* R_x     = (const float*)d_in[4];
    const float* W_delta = (const float*)d_in[5];
    const float* b       = (const float*)d_in[6];
    const float* b_delta = (const float*)d_in[7];
    const float* b_gate  = (const float*)d_in[8];

    float* out  = (float*)d_out;
    float* OUT  = out;                           // output   [T,B,D]
    float* LOG  = out + TBD;                     // log_h    [T+1,B,D]
    float* SGN  = LOG + TP1BD;                   // sign_h   [T+1,B,D]
    float* HLIN = SGN + TP1BD;                   // h_linear [T,B,D]

    char* ws = (char*)d_ws;
    __bf16* xbf  = (__bf16*)(ws + WS_XBF);
    __bf16* rxbf = (__bf16*)(ws + WS_RXBF);
    __bf16* wdbf = (__bf16*)(ws + WS_WDBF);
    __bf16* rhbf = (__bf16*)(ws + WS_RHBF);
    float*  hF0  = (float*)(ws + WS_HF);
    float*  hF1  = hF0 + BD;
    __bf16* hA0  = (__bf16*)(ws + WS_HBF);
    __bf16* hA1  = hA0 + BD;
    float*  u    = (float*)(ws + WS_U);
    float*  v    = (float*)(ws + WS_V);
    float*  wvec = (float*)(ws + WS_W);
    float*  sig  = (float*)(ws + WS_SIGMA);
    unsigned* sync = (unsigned*)(ws + WS_SYNC);

    // 1) bf16 conversions
    cvt_bf16_kernel<<<(int)(TBD/256), 256, 0, stream>>>(x, xbf, (int)TBD);
    cvt_bf16_kernel<<<DD*DD/256, 256, 0, stream>>>(R_x, rxbf, DD*DD);
    cvt_bf16_kernel<<<DD*DD/256, 256, 0, stream>>>(W_delta, wdbf, DD*DD);

    // 2) spectral norm of R_h (3 power iterations), then scale+convert
    u_init_kernel<<<DD/256, 256, 0, stream>>>(u);
    normalize_kernel<<<1, 256, 0, stream>>>(u, 0.0f);
    for (int it = 0; it < 3; ++it) {
        matvecT_kernel<<<DD/256, 256, 0, stream>>>(R_h, u, v);
        normalize_kernel<<<1, 256, 0, stream>>>(v, 1e-8f);
        matvec_kernel<<<DD, 256, 0, stream>>>(R_h, v, u);
        normalize_kernel<<<1, 256, 0, stream>>>(u, 1e-8f);
    }
    matvecT_kernel<<<DD/256, 256, 0, stream>>>(R_h, u, wvec);
    dot_abs_kernel<<<1, 256, 0, stream>>>(wvec, v, sig);
    scale_cvt_rh_kernel<<<DD*DD/256, 256, 0, stream>>>(R_h, sig, rhbf);

    // 3) hoisted GEMMs, staged into d_out slots (overwritten by the scan)
    gemm_pre_kernel<<<4096, 256, 0, stream>>>(xbf, rxbf, b,       OUT);   // P
    gemm_pre_kernel<<<4096, 256, 0, stream>>>(xbf, wdbf, b_delta, HLIN);  // Q

    // 4) init state + t=0 slices + barrier reset
    init_state_kernel<<<BD/256, 256, 0, stream>>>(log_h0, sign_h0, hF0, hA0, LOG, SGN, sync);

    // 5) persistent recurrent scan (32 blocks, 256 KB LDS weight cache each)
    scan_kernel<<<32, 256, 64*DD*2, stream>>>(hA0, hA1, hF0, hF1, rhbf,
                                              OUT, HLIN, LOG + BD, SGN + BD,
                                              x, b_gate, sync);
}
